// Phi4MiniAttention_46437186404653
// MI455X (gfx1250) — compile-verified, matrix operand order checked by reading, not run on hardware
//
#include <hip/hip_runtime.h>

// Phi4MiniAttention for gfx1250 (MI455X): bf16 WMMA GEMMs + flash attention.
// B=2,S=2048,H=3072,NQ=24,NKV=8,HD=128,ROT=96, causal, GQA g=3.
// Compute-bound => v_wmma_f32_16x16x32_bf16 for all matmuls.
// All LDS tiles staged by GLOBAL_LOAD_ASYNC_TO_LDS_B128 (ASYNCcnt) with
// counter-based double buffering: DMA of tile t+1 overlaps WMMA on tile t.
// Operands needing transpose (Wqkv, Wo, V) are pre-transposed once so every
// GEMM/FA tile is K-contiguous and DMA-able.

#define BB   2
#define SS   2048
#define HH   3072
#define NQH  24
#define NKVH 8
#define HD   128
#define NTOK (BB*SS)          // 4096
#define NQKV 5120             // 24*128 + 8*128 + 8*128

typedef __attribute__((ext_vector_type(16))) __bf16    v16bf;
typedef __attribute__((ext_vector_type(8)))  float     v8f;
typedef __attribute__((ext_vector_type(4)))  unsigned  u32x4;

union F8  { v8f v; float f[8]; };

__device__ __forceinline__ unsigned short f2bf(float f) {
  union { float f; unsigned u; } x; x.f = f;
  unsigned r = x.u + 0x7FFFu + ((x.u >> 16) & 1u);   // round-to-nearest-even
  return (unsigned short)(r >> 16);
}

// Async DMA 16B global -> LDS (no VGPR bounce). LDS byte address = low 32 bits
// of the generic pointer (flat aperture truncation rule, ISA 10.2).
__device__ __forceinline__ void async_ld_b128(unsigned short* lds,
                                              const unsigned short* gsrc) {
  unsigned l = (unsigned)(unsigned long long)lds;
  asm volatile("global_load_async_to_lds_b128 %0, %1, off"
               :: "v"(l), "v"(gsrc) : "memory");
}
// Wait until at most N async group-ops remain (in-order completion =>
// everything older than the newest N has landed in LDS).
template <int N>
__device__ __forceinline__ void wait_async() {
  asm volatile("s_wait_asynccnt %0" :: "i"(N) : "memory");
}

// Load one 16x32 bf16 A/B fragment from an LDS tile stored row-major
// ([row][k], stride in ushorts, stride*2 multiple of 16B).
// ISA 16-bit A layout: lane<16 holds k {k0..k0+7, k0+16..k0+23},
// lane>=16 holds k {k0+8..k0+15, k0+24..k0+31}  -> two ds_load_b128.
__device__ __forceinline__ v16bf ldfrag(const unsigned short* p, int row,
                                        int stride, int k0, int hi) {
  const unsigned short* q = p + row * stride + k0 + (hi << 3);
  union { u32x4 u[2]; v16bf v; } r;
  r.u[0] = *(const u32x4*)(q);
  r.u[1] = *(const u32x4*)(q + 16);
  return r.v;
}

__device__ __forceinline__ v8f wmma_bf16(v16bf a, v16bf b, v8f c) {
  return __builtin_amdgcn_wmma_f32_16x16x32_bf16(
      false, a, false, b, (short)0, c, false, false);
}

// Pack two 8-register f32 S^T accumulator tiles into a P A-fragment (bf16).
__device__ __forceinline__ v16bf packP(const float* a, const float* b) {
  union { v16bf v; unsigned short s[16]; } r;
#pragma unroll
  for (int i = 0; i < 8; ++i) { r.s[i] = f2bf(a[i]); r.s[8 + i] = f2bf(b[i]); }
  return r.v;
}

// ---------------- conversion kernels ----------------

__global__ void cvt_f32_bf16(const float* __restrict__ src,
                             unsigned short* __restrict__ dst, int n) {
  int i = blockIdx.x * 256 + threadIdx.x;
  if (i < n) dst[i] = f2bf(src[i]);
}

// Wq[3072][3072],Wk[3072][1024],Wv[3072][1024] -> transposed Wqkv^T [5120][3072]
__global__ void cvt_wqkv_t(const float* __restrict__ Wq, const float* __restrict__ Wk,
                           const float* __restrict__ Wv, unsigned short* __restrict__ W) {
  int i = blockIdx.x * 256 + threadIdx.x;
  if (i >= NQKV * HH) return;
  int n = i / HH, k = i % HH;
  float v;
  if (n < 3072)      v = Wq[(size_t)k * 3072 + n];
  else if (n < 4096) v = Wk[(size_t)k * 1024 + (n - 3072)];
  else               v = Wv[(size_t)k * 1024 + (n - 4096)];
  W[i] = f2bf(v);
}

// dst[n][k] = src[k][n]  (3072 x 3072, for Wo^T)
__global__ void cvt_t_3072(const float* __restrict__ src,
                           unsigned short* __restrict__ dst) {
  int i = blockIdx.x * 256 + threadIdx.x;
  if (i >= HH * HH) return;
  int n = i / HH, k = i % HH;
  dst[i] = f2bf(src[(size_t)k * HH + n]);
}

// ---------------- generic bf16 GEMM:  C[M][N] = A[M][K] * Bt[N][K]^T ----------------
// Both operands K-contiguous => pure async-DMA staging, double-buffered.
// Block tile 128x128, K-step 32, 256 threads = 8 waves (4x2), wave tile 32x64.

#define GLSTR 48   // LDS row stride in ushorts (96B: 16B-aligned, padded)

__global__ __launch_bounds__(256) void gemm_bf16(
    const unsigned short* __restrict__ A, const unsigned short* __restrict__ Bt,
    float* __restrict__ C, int M, int N, int K) {
  __shared__ unsigned short As[2][128 * GLSTR];   // [m][k]
  __shared__ unsigned short Bs[2][128 * GLSTR];   // [n][k]
  const int tid = threadIdx.x;
  const int lane = tid & 31, wave = tid >> 5;
  const int laneLo = lane & 15, hi = (lane >> 4) & 1;
  const int waveM = wave & 3, waveN = wave >> 2;
  const int bm = blockIdx.x * 128, bn = blockIdx.y * 128;
  const int nk = K >> 5;

  // stage tile kt into buffer p: 4 async 16B DMA chunks per thread
  auto stage = [&](int kt, int p) {
    int k0 = kt << 5;
#pragma unroll
    for (int it = 0; it < 2; ++it) {
      int c = tid + it * 256;
      int r = c >> 2, c8 = (c & 3) << 3;
      async_ld_b128(&As[p][r * GLSTR + c8], A  + (size_t)(bm + r) * K + k0 + c8);
      async_ld_b128(&Bs[p][r * GLSTR + c8], Bt + (size_t)(bn + r) * K + k0 + c8);
    }
  };

  F8 acc[2][4];
#pragma unroll
  for (int mt = 0; mt < 2; ++mt)
#pragma unroll
    for (int nt = 0; nt < 4; ++nt)
#pragma unroll
      for (int r = 0; r < 8; ++r) acc[mt][nt].f[r] = 0.0f;

  stage(0, 0);
  for (int kt = 0; kt < nk; ++kt) {
    const int p = kt & 1;
    __syncthreads();                       // readers of buf[1-p] are done
    if (kt + 1 < nk) { stage(kt + 1, 1 - p); wait_async<4>(); }
    else             { wait_async<0>(); }
    __syncthreads();                       // buf[p] fully DMA'd (all waves)

    v16bf af[2], bf[4];
#pragma unroll
    for (int mt = 0; mt < 2; ++mt)
      af[mt] = ldfrag(As[p], waveM * 32 + mt * 16 + laneLo, GLSTR, 0, hi);
#pragma unroll
    for (int nt = 0; nt < 4; ++nt)
      bf[nt] = ldfrag(Bs[p], waveN * 64 + nt * 16 + laneLo, GLSTR, 0, hi);
#pragma unroll
    for (int mt = 0; mt < 2; ++mt)
#pragma unroll
      for (int nt = 0; nt < 4; ++nt)
        acc[mt][nt].v = wmma_bf16(af[mt], bf[nt], acc[mt][nt].v);
  }

  // C/D layout: lane<16 -> col=lane, reg r -> row r; lane>=16 -> row r+8.
#pragma unroll
  for (int mt = 0; mt < 2; ++mt)
#pragma unroll
    for (int nt = 0; nt < 4; ++nt)
#pragma unroll
      for (int r = 0; r < 8; ++r) {
        int row = bm + waveM * 32 + mt * 16 + hi * 8 + r;
        int col = bn + waveN * 64 + nt * 16 + laneLo;
        C[(size_t)row * N + col] = acc[mt][nt].f[r];
      }
}

// ---------------- RoPE (+ bf16 split of QKV; V stored transposed) ----------------
// QKV fp32 [4096][5120]; heads h: 0..23=q, 24..31=k, 32..39=v, col base = h*128.
// V goes out as Vt[b][kvh][hd][s] so flash-attn can DMA it key-contiguous.

__global__ void rope_split(const float* __restrict__ QKV,
                           unsigned short* __restrict__ Qb,
                           unsigned short* __restrict__ Kb,
                           unsigned short* __restrict__ Vt) {
  int gid = blockIdx.x * 256 + threadIdx.x;
  if (gid >= NTOK * 40 * 64) return;
  int i = gid & 63;
  int t2 = gid >> 6;
  int h = t2 % 40, t = t2 / 40;
  int s = t & (SS - 1);
  const float* row = QKV + (size_t)t * NQKV + h * 128;

  if (h >= 32) {                       // V: transpose-store, 2 elems/thread
    int bb = t >> 11;
    size_t base = (size_t)(bb * NKVH + (h - 32)) * HD;
    Vt[(base + 2 * i)     * SS + s] = f2bf(row[2 * i]);
    Vt[(base + 2 * i + 1) * SS + s] = f2bf(row[2 * i + 1]);
    return;
  }
  unsigned short* out = (h < 24)
      ? Qb + (size_t)t * (NQH * HD) + h * 128
      : Kb + (size_t)t * (NKVH * HD) + (h - 24) * 128;
  if (i < 48) {                        // rotary pair within first 96 dims
    float inv_freq = __expf(-(float)i * (1.0f / 48.0f) * 9.210340371976184f); // ln(1e4)
    float fr = (float)s * inv_freq;
    float c = __cosf(fr), sn = __sinf(fr);
    float x1 = row[i], x2 = row[i + 48];
    out[i]      = f2bf(x1 * c - x2 * sn);
    out[i + 48] = f2bf(x2 * c + x1 * sn);
  } else {                             // pass-through dims 96..127
    int j = 96 + ((i - 48) << 1);
    out[j]     = f2bf(row[j]);
    out[j + 1] = f2bf(row[j + 1]);
  }
}

// ---------------- flash attention ----------------
// One WG = (b, q-head u, 128-query tile). 8 waves x 16 queries each.
// S^T = K(64x128) @ Q^T  -> lane = query, regs = keys  (softmax in-register).
// S^T accum tiles map directly onto P A-fragments for O += P @ V.
// K/V tiles DMA'd async, double-buffered; V pre-transposed in global.

#define QTILE 128
#define KTILE 64
#define QSTR  136   // 272B rows: 16B aligned
#define KSTR  136
#define VSTR  72    // 144B rows: 16B aligned

__global__ __launch_bounds__(256) void flash_attn(
    const unsigned short* __restrict__ Qb, const unsigned short* __restrict__ Kb,
    const unsigned short* __restrict__ Vt, unsigned short* __restrict__ Ob) {
  __shared__ unsigned short Qs[QTILE * QSTR];     // [query][hd]
  __shared__ unsigned short Ks[2][KTILE * KSTR];  // [key][hd]
  __shared__ unsigned short Vs[2][HD * VSTR];     // [hd][key]

  const int tid = threadIdx.x;
  const int lane = tid & 31, wave = tid >> 5;
  const int laneLo = lane & 15, hi = (lane >> 4) & 1;

  int idx = blockIdx.x;
  const int qt = idx % (SS / QTILE); idx /= (SS / QTILE);
  const int u  = idx % NQH;
  const int b  = idx / NQH;
  const int kvh = u / 3;                      // GQA: repeat(k,3) -> u//3
  const size_t qrow0 = (size_t)b * SS + (size_t)qt * QTILE;
  const size_t vrow0 = (size_t)(b * NKVH + kvh) * HD;

  // stage K/V tile kb into buffer p: 8 async 16B DMA chunks per thread
  auto stageKV = [&](int kb, int p) {
    int kv0 = kb * KTILE;
#pragma unroll
    for (int it = 0; it < 4; ++it) {          // K: 64 x 128, hd-contiguous
      int c = tid + it * 256;
      int r = c >> 4, h8 = (c & 15) << 3;
      async_ld_b128(&Ks[p][r * KSTR + h8],
                    Kb + ((size_t)b * SS + kv0 + r) * (NKVH * HD) + kvh * 128 + h8);
    }
#pragma unroll
    for (int it = 0; it < 4; ++it) {          // V^T: 128 x 64, key-contiguous
      int c = tid + it * 256;
      int hd = c >> 3, k8 = (c & 7) << 3;
      async_ld_b128(&Vs[p][hd * VSTR + k8],
                    Vt + (vrow0 + hd) * SS + kv0 + k8);
    }
  };

  // stage Q block: 2048 async 16B DMA chunks, 8 per thread
#pragma unroll
  for (int it = 0; it < 8; ++it) {
    int c = tid + it * 256;
    int r = c >> 4, h8 = (c & 15) << 3;
    async_ld_b128(&Qs[r * QSTR + h8],
                  Qb + (qrow0 + r) * (NQH * HD) + u * 128 + h8);
  }

  F8 accO[8];
#pragma unroll
  for (int nt = 0; nt < 8; ++nt)
#pragma unroll
    for (int r = 0; r < 8; ++r) accO[nt].f[r] = 0.0f;
  float m = -1.0e30f, l = 0.0f;
  const int qglob = qt * QTILE + wave * 16 + laneLo;   // this lane's query
  const float inv_norm = 0.08838834764831845f;          // 1/sqrt(128)
  const int nKV = 2 * qt + 2;                           // causal block count

  stageKV(0, 0);
  for (int kb = 0; kb < nKV; ++kb) {
    const int kv0 = kb * KTILE;
    const int p = kb & 1;
    __syncthreads();                       // readers of buf[1-p] are done
    if (kb + 1 < nKV) { stageKV(kb + 1, 1 - p); wait_async<8>(); }
    else              { wait_async<0>(); }
    __syncthreads();                       // buf[p] (and Qs) fully DMA'd

    // S^T: 4 key m-tiles x 4 hd k-steps
    F8 sacc[4];
#pragma unroll
    for (int mt = 0; mt < 4; ++mt)
#pragma unroll
      for (int r = 0; r < 8; ++r) sacc[mt].f[r] = 0.0f;
#pragma unroll
    for (int kk = 0; kk < 4; ++kk) {
      v16bf qf = ldfrag(Qs, wave * 16 + laneLo, QSTR, kk * 32, hi);
#pragma unroll
      for (int mt = 0; mt < 4; ++mt) {
        v16bf kf = ldfrag(Ks[p], mt * 16 + laneLo, KSTR, kk * 32, hi);
        sacc[mt].v = wmma_bf16(kf, qf, sacc[mt].v);
      }
    }

    // online softmax (row = query = this lane; combine lane pairs via xor16)
    float mloc = -1.0e30f;
#pragma unroll
    for (int mt = 0; mt < 4; ++mt)
#pragma unroll
      for (int r = 0; r < 8; ++r) {
        int key = kv0 + mt * 16 + hi * 8 + r;
        float vv = sacc[mt].f[r] * inv_norm;
        if (key > qglob) vv = -1.0e30f;                 // causal mask
        sacc[mt].f[r] = vv;
        mloc = fmaxf(mloc, vv);
      }
    mloc = fmaxf(mloc, __shfl_xor(mloc, 16));
    float mnew  = fmaxf(m, mloc);
    float alpha = __expf(m - mnew);
    float lloc = 0.0f;
#pragma unroll
    for (int mt = 0; mt < 4; ++mt)
#pragma unroll
      for (int r = 0; r < 8; ++r) {
        float pv = __expf(sacc[mt].f[r] - mnew);
        sacc[mt].f[r] = pv;
        lloc += pv;
      }
    lloc += __shfl_xor(lloc, 16);
    l = l * alpha + lloc;
    m = mnew;
    // rescale O; factor for query q=r+hi*8 lives in lane q
#pragma unroll
    for (int r = 0; r < 8; ++r) {
      float a = __shfl(alpha, r + hi * 8, 32);
#pragma unroll
      for (int nt = 0; nt < 8; ++nt) accO[nt].f[r] *= a;
    }
    // O += P @ V  (P A-frags come straight out of sacc registers)
#pragma unroll
    for (int ks = 0; ks < 2; ++ks) {
      v16bf pf = packP(sacc[2 * ks].f, sacc[2 * ks + 1].f);
#pragma unroll
      for (int nt = 0; nt < 8; ++nt) {
        v16bf vf = ldfrag(Vs[p], nt * 16 + laneLo, VSTR, ks * 32, hi);
        accO[nt].v = wmma_bf16(pf, vf, accO[nt].v);
      }
    }
  }

  // normalize and store bf16 [token][u*128+hd]
#pragma unroll
  for (int r = 0; r < 8; ++r) {
    float il = 1.0f / __shfl(l, r + hi * 8, 32);
    size_t row = qrow0 + wave * 16 + hi * 8 + r;
#pragma unroll
    for (int nt = 0; nt < 8; ++nt) {
      int col = u * 128 + nt * 16 + laneLo;
      Ob[row * (NQH * HD) + col] = f2bf(accO[nt].f[r] * il);
    }
  }
}

// ---------------- host launcher ----------------

extern "C" void kernel_launch(void* const* d_in, const int* in_sizes, int n_in,
                              void* d_out, int out_size, void* d_ws, size_t ws_size,
                              hipStream_t stream) {
  const float* hidden = (const float*)d_in[0];
  // d_in[1] = attention_mask (bool causal) -- implemented analytically, unused
  const float* Wq = (const float*)d_in[2];
  const float* Wk = (const float*)d_in[3];
  const float* Wv = (const float*)d_in[4];
  const float* Wo = (const float*)d_in[5];
  float* out = (float*)d_out;

  char* ws = (char*)d_ws;
  unsigned short* Xbf   = (unsigned short*)(ws);                 // 4096x3072 bf16
  unsigned short* Wqkvb = (unsigned short*)(ws + 25165824);      // 5120x3072 bf16 (T)
  unsigned short* Wob   = (unsigned short*)(ws + 56623104);      // 3072x3072 bf16 (T)
  float*          QKVf  = (float*)        (ws + 75497472);       // 4096x5120 f32
  unsigned short* Qbf   = (unsigned short*)(ws + 159383552);     // 4096x3072 bf16
  unsigned short* Kbf   = (unsigned short*)(ws + 184549376);     // 4096x1024 bf16
  unsigned short* Vtb   = (unsigned short*)(ws + 192937984);     // [2][8][128][2048] bf16
  unsigned short* Attnb = (unsigned short*)(ws + 201326592);     // 4096x3072 bf16
  (void)ws_size; (void)in_sizes; (void)n_in; (void)out_size;     // ~216MB used

  // 1) conversions (weights pre-transposed for K-contiguous DMA)
  cvt_f32_bf16<<<(NTOK * HH + 255) / 256, 256, 0, stream>>>(hidden, Xbf, NTOK * HH);
  cvt_wqkv_t<<<(NQKV * HH + 255) / 256, 256, 0, stream>>>(Wq, Wk, Wv, Wqkvb);
  cvt_t_3072<<<(HH * HH + 255) / 256, 256, 0, stream>>>(Wo, Wob);

  // 2) QKV projection: [4096x3072] @ [3072x5120] -> f32
  gemm_bf16<<<dim3(NTOK / 128, NQKV / 128), 256, 0, stream>>>(
      Xbf, Wqkvb, QKVf, NTOK, NQKV, HH);

  // 3) RoPE + split to bf16 Q/K (row-major) and V (transposed)
  rope_split<<<(NTOK * 40 * 64 + 255) / 256, 256, 0, stream>>>(QKVf, Qbf, Kbf, Vtb);

  // 4) causal GQA flash attention
  flash_attn<<<BB * NQH * (SS / QTILE), 256, 0, stream>>>(Qbf, Kbf, Vtb, Attnb);

  // 5) output projection: [4096x3072] @ [3072x3072] -> f32 out
  gemm_bf16<<<dim3(NTOK / 128, HH / 128), 256, 0, stream>>>(
      Attnb, Wob, out, NTOK, HH, HH);
}
